// GlobalContext_19516331393320
// MI455X (gfx1250) — matrix operand order repaired and checked by reading, not compiled, not run on hardware
//
#include <hip/hip_runtime.h>
#include <hip/hip_bf16.h>

// ---------------------------------------------------------------------------
// MI455X (gfx1250) fused GAT-style attention, v3.
// Bandwidth-bound: bias_mat read x2 (819MB) + coefs write (410MB) dominate
// (~58us @ 23.3TB/s). v3 changes:
//  * W1 and Wsum pre-packed to f16 WMMA B-fragment order (one-time kernels)
//    -> k_uhf/k_final inner loops are pure v16h (b128) loads + wmma.
//  * k_f2 vectorized (f4).
// Retained from v2: G=8 N-split (512 WGs in heavy kernels), uhf_h f16
// fragment repack, NT b128 streaming for bias/coefs, deterministic
// per-segment vals partials (no FP atomics).
// ---------------------------------------------------------------------------

typedef __attribute__((ext_vector_type(16))) _Float16 v16h;
typedef __attribute__((ext_vector_type(8)))  float    v8f;
typedef __attribute__((ext_vector_type(4)))  float    f4;

#define NN 100000   // nodes
#define FF 128      // input features
#define BB 1024     // batch
#define KFF 128     // knowledge features
#define UU 64       // hidden unit
#define HH 4        // gat heads (identical weights -> folded into Wsum)
#define G  8        // N-dimension split for the heavy kernels
#define NCH (NN/32) // 3125 chunks of 32 nodes

__device__ __forceinline__ v8f wmma_f16(v16h a, v16h b, v8f c) {
  return __builtin_amdgcn_wmma_f32_16x16x32_f16(
      false, a, false, b, (short)0, c, false, false);
}

// ---------------------------------------------------------------------------
// Pack W1 [128,64] f32 -> f16 B-fragment order: frag (s,ut) covers
// K=[s*32,+32), U=[ut*16,+16). 512 threads, one v16h store each.
// ---------------------------------------------------------------------------
__global__ void k_packW1(const float* __restrict__ W1, _Float16* __restrict__ W1h) {
  const int t = blockIdx.x * blockDim.x + threadIdx.x;
  if (t >= 512) return;                       // 4 ksteps * 4 utiles * 32 lanes
  const int s  = t >> 7;
  const int ut = (t >> 5) & 3;
  const int l  = t & 31;
  const int u  = ut * 16 + (l & 15);
  const int k0 = s * 32 + (l >> 4) * 8;
  v16h v;
#pragma unroll
  for (int e = 0; e < 16; ++e) v[e] = (_Float16)W1[(k0 + (e < 8 ? e : e + 8)) * UU + u];
  ((v16h*)W1h)[t] = v;
}

// ---------------------------------------------------------------------------
// Wsum = sum_h Wout[h*64:(h+1)*64,:], emitted directly in f16 B-frag order.
// 256 threads (2 ksteps * 4 utiles * 32 lanes).
// ---------------------------------------------------------------------------
__global__ void k_wsumh(const float* __restrict__ Wout, _Float16* __restrict__ Wsumh) {
  const int t = blockIdx.x * blockDim.x + threadIdx.x;
  if (t >= 256) return;
  const int s  = t >> 7;
  const int ut = (t >> 5) & 3;
  const int l  = t & 31;
  const int u  = ut * 16 + (l & 15);
  const int k0 = s * 32 + (l >> 4) * 8;
  v16h v;
#pragma unroll
  for (int e = 0; e < 16; ++e) {
    float sum = 0.f;
#pragma unroll
    for (int h = 0; h < HH; ++h) sum += Wout[(h * UU + k0 + (e < 8 ? e : e + 8)) * UU + u];
    v[e] = (_Float16)sum;
  }
  ((v16h*)Wsumh)[t] = v;
}

// ---------------------------------------------------------------------------
// Kernel: uhf = x @ W1   [100000,128]@[128,64] -> f32. 8 waves/block,
// 16 rows/wave. A from x via 4x b128 + cvt; B from packed W1h via v16h.
// ---------------------------------------------------------------------------
__global__ __launch_bounds__(256) void k_uhf(const float* __restrict__ x,
                                             const _Float16* __restrict__ W1h,
                                             float* __restrict__ uhf) {
  const int wave  = threadIdx.x >> 5;
  const int lane  = threadIdx.x & 31;
  const int rowb  = (blockIdx.x * 8 + wave) * 16;
  const int r     = lane & 15;
  const int khalf = lane >> 4;
  int arow = rowb + r;
  if (arow >= NN) arow = NN - 1;
  const v16h* W1v = (const v16h*)W1h;

  v8f c[4] = {};
  for (int s = 0; s < 4; ++s) {
    const float* xb = x + arow * FF + s * 32 + khalf * 8;
    f4 a0 = *(const f4*)(xb);
    f4 a1 = *(const f4*)(xb + 4);
    f4 a2 = *(const f4*)(xb + 16);
    f4 a3 = *(const f4*)(xb + 20);
    v16h a;
#pragma unroll
    for (int j = 0; j < 4; ++j) {
      a[j]      = (_Float16)a0[j];
      a[j + 4]  = (_Float16)a1[j];
      a[j + 8]  = (_Float16)a2[j];
      a[j + 12] = (_Float16)a3[j];
    }
#pragma unroll
    for (int ut = 0; ut < 4; ++ut)
      c[ut] = wmma_f16(a, W1v[(s * 4 + ut) * 32 + lane], c[ut]);
  }
#pragma unroll
  for (int ut = 0; ut < 4; ++ut)
#pragma unroll
    for (int rr = 0; rr < 8; ++rr) {
      const int m = rowb + rr + khalf * 8;
      if (m < NN) uhf[m * UU + ut * 16 + r] = c[ut][rr];
    }
}

// ---------------------------------------------------------------------------
// Repack uhf f32 -> f16 in exact B-fragment order for the coefs@uhf GEMM.
// ---------------------------------------------------------------------------
__global__ void k_pack(const float* __restrict__ uhf, _Float16* __restrict__ uhf_h) {
  const int t = blockIdx.x * blockDim.x + threadIdx.x;
  if (t >= NCH * 128) return;
  const int ch = t >> 7;
  const int ut = (t >> 5) & 3;
  const int l  = t & 31;
  const int u  = ut * 16 + (l & 15);
  const int n0 = ch * 32 + (l >> 4) * 8;
  v16h v;
#pragma unroll
  for (int e = 0; e < 16; ++e) v[e] = (_Float16)uhf[(n0 + (e < 8 ? e : e + 8)) * UU + u];
  ((v16h*)uhf_h)[t] = v;
}

// ---------------------------------------------------------------------------
// f2[n] = uhf[n,:] . W3 + b3   (f4-vectorized)
// ---------------------------------------------------------------------------
__global__ void k_f2(const float* __restrict__ uhf, const float* __restrict__ W3,
                     const float* __restrict__ b3, float* __restrict__ f2) {
  const int n = blockIdx.x * blockDim.x + threadIdx.x;
  if (n >= NN) return;
  const f4* row = (const f4*)(uhf + n * UU);
  const f4* w   = (const f4*)W3;
  float acc = b3[0];
#pragma unroll
  for (int q = 0; q < UU / 4; ++q) {
    const f4 a = row[q], b = w[q];
    acc += a[0] * b[0] + a[1] * b[1] + a[2] * b[2] + a[3] * b[3];
  }
  f2[n] = acc;
}

// ---------------------------------------------------------------------------
// f1[b] = (relu(ukg[b]@Wk + bk) + uhf[user_index[b]]) . W2 + b2
// ---------------------------------------------------------------------------
__global__ __launch_bounds__(64) void k_f1(const float* __restrict__ ukg,
                                           const float* __restrict__ Wk,
                                           const float* __restrict__ bk,
                                           const float* __restrict__ W2,
                                           const float* __restrict__ b2,
                                           const int*   __restrict__ user_index,
                                           const float* __restrict__ uhf,
                                           float* __restrict__ f1) {
  __shared__ float red[64];
  const int b = blockIdx.x;
  const int u = threadIdx.x;
  float acc = bk[u];
  for (int k = 0; k < KFF; ++k) acc += ukg[b * KFF + k] * Wk[k * UU + u];
  float bh = acc > 0.f ? acc : 0.f;
  red[u] = (bh + uhf[user_index[b] * UU + u]) * W2[u];
  __syncthreads();
  for (int off = 32; off > 0; off >>= 1) {
    if (u < off) red[u] += red[u + off];
    __syncthreads();
  }
  if (u == 0) f1[b] = red[0] + b2[0];
}

__device__ __forceinline__ float logit_of(float f1v, float f2v, float bias) {
  float l = f1v + f2v;
  l = l > 0.f ? l : 0.2f * l;          // leaky_relu(0.2)
  return l - 1e9f * (1.0f - bias);     // + adj
}

// ---------------------------------------------------------------------------
// Per-(row, segment) online softmax stats. Grid (B/16, G), 256 threads.
// ---------------------------------------------------------------------------
__global__ __launch_bounds__(256) void k_stats(const float* __restrict__ bias_mat,
                                               const float* __restrict__ f1g,
                                               const float* __restrict__ f2g,
                                               float* __restrict__ statsM,
                                               float* __restrict__ statsS) {
  __shared__ float sM[256], sS[256];
  const int tid  = threadIdx.x;
  const int r    = tid >> 4;
  const int c    = tid & 15;
  const int brow = blockIdx.x * 16 + r;
  const int seg  = blockIdx.y;
  const float f1v = f1g[brow];

  const f4* biasq = (const f4*)(bias_mat + (size_t)brow * NN);
  const f4* f2q   = (const f4*)f2g;
  const int q0 = seg * (NN / 4 / G);
  const int q1 = q0 + (NN / 4 / G);

  float m = -3.4e38f, s = 0.f;
  for (int q = q0 + c; q < q1; q += 16) {
    const f4 bv = __builtin_nontemporal_load(biasq + q);
    const f4 fv = f2q[q];
#pragma unroll
    for (int j = 0; j < 4; ++j) {
      const float l  = logit_of(f1v, fv[j], bv[j]);
      const float mn = fmaxf(m, l);
      s = s * __expf(m - mn) + __expf(l - mn);
      m = mn;
    }
  }
  sM[tid] = m; sS[tid] = s;
  __syncthreads();
  if (c == 0) {
    for (int j = 1; j < 16; ++j) {
      const float m2 = sM[tid + j], s2 = sS[tid + j];
      const float mn = fmaxf(m, m2);
      s = s * __expf(m - mn) + s2 * __expf(m2 - mn);
      m = mn;
    }
    statsM[brow * G + seg] = m;
    statsS[brow * G + seg] = s;
  }
}

// ---------------------------------------------------------------------------
// Combine segment stats -> rowM, rowIS
// ---------------------------------------------------------------------------
__global__ void k_rowstat(const float* __restrict__ statsM,
                          const float* __restrict__ statsS,
                          float* __restrict__ rowM, float* __restrict__ rowIS) {
  const int b = blockIdx.x * blockDim.x + threadIdx.x;
  if (b >= BB) return;
  float m = statsM[b * G], s = statsS[b * G];
#pragma unroll
  for (int g = 1; g < G; ++g) {
    const float m2 = statsM[b * G + g], s2 = statsS[b * G + g];
    const float mn = fmaxf(m, m2);
    s = s * __expf(m - mn) + s2 * __expf(m2 - mn);
    m = mn;
  }
  rowM[b] = m;
  rowIS[b] = 1.0f / s;
}

// ---------------------------------------------------------------------------
// coefs + vals partials. Grid (B/16, G), 8 waves. Per chunk:
// 4x b128 NT bias load, 4x b128 f2 load, 4x b128 NT coef store,
// 4 WMMAs against pre-swizzled uhf_h (one v16h = 2x b128 each, L2-hit).
// ---------------------------------------------------------------------------
__global__ __launch_bounds__(256) void k_coef(const float* __restrict__ bias_mat,
                                              const _Float16* __restrict__ uhf_h,
                                              const float* __restrict__ f1g,
                                              const float* __restrict__ f2g,
                                              const float* __restrict__ rowMg,
                                              const float* __restrict__ rowISg,
                                              float* __restrict__ coefs,
                                              float* __restrict__ valsSeg) {
  __shared__ float valsP[8 * 16 * 64];

  const int tid   = threadIdx.x;
  const int wave  = tid >> 5;
  const int lane  = tid & 31;
  const int r     = lane & 15;
  const int khalf = lane >> 4;
  const int btile = blockIdx.x;
  const int seg   = blockIdx.y;
  const int brow  = btile * 16 + r;
  const float f1v  = f1g[brow];
  const float mrow = rowMg[brow];
  const float isrw = rowISg[brow];
  const v16h* uhv  = (const v16h*)uhf_h;

  v8f cacc[4] = {};
  for (int ch = seg * 8 + wave; ch < NCH; ch += G * 8) {
    const size_t base = (size_t)brow * NN + ch * 32 + khalf * 8;
    const int fb = ch * 32 + khalf * 8;
    const f4 b0 = __builtin_nontemporal_load((const f4*)(bias_mat + base));
    const f4 b1 = __builtin_nontemporal_load((const f4*)(bias_mat + base + 4));
    const f4 b2 = __builtin_nontemporal_load((const f4*)(bias_mat + base + 16));
    const f4 b3 = __builtin_nontemporal_load((const f4*)(bias_mat + base + 20));
    const f4 f0  = *(const f4*)(f2g + fb);
    const f4 f1_ = *(const f4*)(f2g + fb + 4);
    const f4 f2_ = *(const f4*)(f2g + fb + 16);
    const f4 f3  = *(const f4*)(f2g + fb + 20);
    f4 c0, c1, c2, c3;
    v16h a;
#pragma unroll
    for (int j = 0; j < 4; ++j) {
      c0[j] = __expf(logit_of(f1v, f0[j],  b0[j]) - mrow) * isrw;
      c1[j] = __expf(logit_of(f1v, f1_[j], b1[j]) - mrow) * isrw;
      c2[j] = __expf(logit_of(f1v, f2_[j], b2[j]) - mrow) * isrw;
      c3[j] = __expf(logit_of(f1v, f3[j],  b3[j]) - mrow) * isrw;
      a[j]      = (_Float16)c0[j];
      a[j + 4]  = (_Float16)c1[j];
      a[j + 8]  = (_Float16)c2[j];
      a[j + 12] = (_Float16)c3[j];
    }
    __builtin_nontemporal_store(c0, (f4*)(coefs + base));
    __builtin_nontemporal_store(c1, (f4*)(coefs + base + 4));
    __builtin_nontemporal_store(c2, (f4*)(coefs + base + 16));
    __builtin_nontemporal_store(c3, (f4*)(coefs + base + 20));
#pragma unroll
    for (int ut = 0; ut < 4; ++ut)
      cacc[ut] = wmma_f16(a, uhv[(ch * 4 + ut) * 32 + lane], cacc[ut]);
  }
#pragma unroll
  for (int ut = 0; ut < 4; ++ut)
#pragma unroll
    for (int rr = 0; rr < 8; ++rr)
      valsP[wave * 1024 + (rr + khalf * 8) * 64 + ut * 16 + r] = cacc[ut][rr];
  __syncthreads();

  float* dst = valsSeg + ((size_t)seg * BB + btile * 16) * UU;
  for (int i = tid; i < 1024; i += 256) {
    float s = 0.f;
#pragma unroll
    for (int w = 0; w < 8; ++w) s += valsP[w * 1024 + i];
    dst[i] = s;
  }
}

// ---------------------------------------------------------------------------
// vals = sum_seg partials; ret = relu(vals + b); out = relu(ret @ Wsum).
// One wave per 16-row tile; B from pre-packed Wsumh (v16h loads).
// ---------------------------------------------------------------------------
__global__ __launch_bounds__(32) void k_final(const float* __restrict__ valsSeg,
                                              const float* __restrict__ bvec,
                                              const _Float16* __restrict__ Wsumh,
                                              float* __restrict__ outp) {
  const int lane  = threadIdx.x;
  const int r     = lane & 15;
  const int khalf = lane >> 4;
  const int bbase = blockIdx.x * 16;
  const int brow  = bbase + r;
  const v16h* Wv  = (const v16h*)Wsumh;

  v8f co[4] = {};
  for (int s = 0; s < 2; ++s) {
    v16h a;
#pragma unroll
    for (int e = 0; e < 16; ++e) {
      const int k = s * 32 + khalf * 8 + (e < 8 ? e : e + 8);
      float v = bvec[k];
#pragma unroll
      for (int g = 0; g < G; ++g) v += valsSeg[(size_t)g * BB * UU + brow * UU + k];
      a[e] = (_Float16)(v > 0.f ? v : 0.f);
    }
#pragma unroll
    for (int ut = 0; ut < 4; ++ut)
      co[ut] = wmma_f16(a, Wv[(s * 4 + ut) * 32 + lane], co[ut]);
  }
#pragma unroll
  for (int ut = 0; ut < 4; ++ut)
#pragma unroll
    for (int rr = 0; rr < 8; ++rr) {
      const float v = co[ut][rr];
      outp[(bbase + rr + khalf * 8) * UU + ut * 16 + r] = v > 0.f ? v : 0.f;
    }
}

// ---------------------------------------------------------------------------
extern "C" void kernel_launch(void* const* d_in, const int* in_sizes, int n_in,
                              void* d_out, int out_size, void* d_ws, size_t ws_size,
                              hipStream_t stream) {
  const float* x        = (const float*)d_in[0];
  const float* bias_mat = (const float*)d_in[1];
  const float* ukg      = (const float*)d_in[2];
  const int*   uidx     = (const int*)  d_in[3];
  const float* W1       = (const float*)d_in[4];
  const float* Wk       = (const float*)d_in[5];
  const float* bk       = (const float*)d_in[6];
  const float* W2       = (const float*)d_in[7];
  const float* b2       = (const float*)d_in[8];
  const float* W3       = (const float*)d_in[9];
  const float* b3       = (const float*)d_in[10];
  const float* bvec     = (const float*)d_in[11];
  const float* Wout     = (const float*)d_in[12];
  (void)in_sizes; (void)n_in; (void)out_size; (void)ws_size;

  // workspace layout (32B-aligned slabs), ~42 MB total
  float*     ws      = (float*)d_ws;
  float*     uhf     = ws;                                   // N*64 f32
  _Float16*  uhf_h   = (_Float16*)(uhf + (size_t)NN * UU);   // N*64 f16 (frag order)
  float*     f2      = (float*)(uhf_h + (size_t)NN * UU);    // N
  float*     f1      = f2 + NN;                              // B
  _Float16*  W1h     = (_Float16*)(f1 + BB);                 // 128*64 f16 (frag order)
  _Float16*  Wsumh   = W1h + FF * UU;                        // 64*64 f16 (frag order)
  float*     statsM  = (float*)(Wsumh + UU * UU);            // B*G
  float*     statsS  = statsM + BB * G;                      // B*G
  float*     rowM    = statsS + BB * G;                      // B
  float*     rowIS   = rowM + BB;                            // B
  float*     valsSeg = rowIS + BB;                           // G*B*64

  float* outp  = (float*)d_out;
  float* coefs = outp + (size_t)BB * UU;

  k_packW1 <<<2, 256, 0, stream>>>(W1, W1h);
  k_wsumh  <<<1, 256, 0, stream>>>(Wout, Wsumh);
  k_uhf    <<<(NN + 127) / 128, 256, 0, stream>>>(x, W1h, uhf);
  k_pack   <<<(NCH * 128 + 255) / 256, 256, 0, stream>>>(uhf, uhf_h);
  k_f2     <<<(NN + 255) / 256, 256, 0, stream>>>(uhf, W3, b3, f2);
  k_f1     <<<BB, 64, 0, stream>>>(ukg, Wk, bk, W2, b2, uidx, uhf, f1);
  k_stats  <<<dim3(BB / 16, G), 256, 0, stream>>>(bias_mat, f1, f2, statsM, statsS);
  k_rowstat<<<(BB + 255) / 256, 256, 0, stream>>>(statsM, statsS, rowM, rowIS);
  k_coef   <<<dim3(BB / 16, G), 256, 0, stream>>>(bias_mat, uhf_h, f1, f2,
                                                  rowM, rowIS, coefs, valsSeg);
  k_final  <<<BB / 16, 32, 0, stream>>>(valsSeg, bvec, Wsumh, outp);
}
